// GCN_39195871543847
// MI455X (gfx1250) — compile-verified, matrix operand order checked by reading, not run on hardware
//
#include <hip/hip_runtime.h>
#include <hip/hip_bf16.h>

#define N_NODES 10000
#define E_EDGES 160000
#define F_INDIM 2208
#define HDIM    512
#define CDIM    2

#define BM 128
#define BN 128
#define BK 32
#define LDSROW 40   // BK + 8 bf16 pad -> 80B row stride, conflict-free ds_load_b128
#define MPAD 10112  // 79 * 128

typedef __attribute__((ext_vector_type(16))) __bf16 v16bf;
typedef __attribute__((ext_vector_type(8)))  float  v8f;

union BF16x16 { v16bf v; uint4 q[2]; };

// ---------------- graph normalization ----------------
__global__ void k_deg_init(float* __restrict__ deg) {
    int i = blockIdx.x * blockDim.x + threadIdx.x;
    if (i < N_NODES) deg[i] = 1.0f;   // self-loop contributes 1
}

__global__ void k_deg_edges(const int* __restrict__ dst, float* __restrict__ deg) {
    int e = blockIdx.x * blockDim.x + threadIdx.x;
    if (e < E_EDGES) unsafeAtomicAdd(&deg[dst[e]], 1.0f);
}

__global__ void k_dinv(const float* __restrict__ deg, float* __restrict__ dinv) {
    int i = blockIdx.x * blockDim.x + threadIdx.x;
    if (i < N_NODES) {
        float d = deg[i];
        dinv[i] = d > 0.0f ? rsqrtf(d) : 0.0f;
    }
}

__global__ void k_norm(const int* __restrict__ src, const int* __restrict__ dst,
                       const float* __restrict__ dinv, float* __restrict__ norm) {
    int e = blockIdx.x * blockDim.x + threadIdx.x;
    if (e < E_EDGES) norm[e] = dinv[src[e]] * dinv[dst[e]];
}

// ---------------- precision conversion ----------------
__global__ void k_cast_bf16(const float* __restrict__ in, __hip_bfloat16* __restrict__ out,
                            size_t n) {
    size_t i = (size_t)blockIdx.x * blockDim.x + threadIdx.x;
    if (i < n) out[i] = __float2bfloat16(in[i]);
}

// in: K x Nd row-major (f32) -> out: Nd x K row-major (bf16)
__global__ void k_transpose_bf16(const float* __restrict__ in, __hip_bfloat16* __restrict__ out,
                                 int K, int Nd) {
    int i = blockIdx.x * blockDim.x + threadIdx.x;
    if (i < K * Nd) {
        int k = i / Nd, n = i % Nd;
        out[(size_t)n * K + k] = __float2bfloat16(in[i]);
    }
}

// ---------------- WMMA GEMM with async-LDS double buffering ----------------
// C[MPAD,H] = A[MPAD,K](bf16) @ Bt[H,K](bf16)^T
// Block: 256 thr (8 waves), 128x128 tile. Wave: 32x64 (2x4 WMMA accumulators).
// A/B tiles staged to LDS with global_load_async_to_lds_b128 (ASYNCcnt),
// double-buffered: wait asynccnt<=4 keeps the next stage's 4 per-wave asyncs in flight.
template <int K>
__global__ __launch_bounds__(256) void k_gemm_lds(
        const __hip_bfloat16* __restrict__ A,
        const __hip_bfloat16* __restrict__ Bt,
        float* __restrict__ C, int Hout) {
    __shared__ __align__(16) __hip_bfloat16 As[2][BM][LDSROW];
    __shared__ __align__(16) __hip_bfloat16 Bs[2][BN][LDSROW];

    const int tid   = threadIdx.x;
    const int mBase = blockIdx.y * BM;
    const int nBase = blockIdx.x * BN;
    const int wave  = tid >> 5;
    const int lane  = tid & 31;
    const int waveM = wave >> 1;      // 0..3 -> 32-row strip
    const int waveN = wave & 1;       // 0..1 -> 64-col strip
    const int laneM = lane & 15;
    const int laneK = (lane >> 4) << 3;   // ISA 16-bit A/B layout: hi lanes hold K+8/K+24

    // stage-copy mapping: 2 chunks of 16B per thread per tile
    const int crow = tid >> 2;            // rows 0..63 (i=0), 64..127 (i=1)
    const int ckq  = (tid & 3) << 3;      // K element offset 0/8/16/24

    auto issue = [&](int buf, int k0) {
        const __hip_bfloat16* Ag = A  + (size_t)mBase * K + k0;
        const __hip_bfloat16* Bg = Bt + (size_t)nBase * K + k0;
#pragma unroll
        for (int i = 0; i < 2; ++i) {
            int row = crow + (i << 6);
            unsigned la = (unsigned)(uintptr_t)&As[buf][row][ckq];
            const void* ga = (const void*)(Ag + (size_t)row * K + ckq);
            asm volatile("global_load_async_to_lds_b128 %0, %1, off"
                         :: "v"(la), "v"(ga) : "memory");
            unsigned lb = (unsigned)(uintptr_t)&Bs[buf][row][ckq];
            const void* gb = (const void*)(Bg + (size_t)row * K + ckq);
            asm volatile("global_load_async_to_lds_b128 %0, %1, off"
                         :: "v"(lb), "v"(gb) : "memory");
        }
    };

    constexpr int STAGES = K / BK;
    v8f acc[2][4];
#pragma unroll
    for (int mf = 0; mf < 2; ++mf)
#pragma unroll
        for (int nf = 0; nf < 4; ++nf) acc[mf][nf] = v8f{};

    issue(0, 0);
    for (int s = 0; s < STAGES; ++s) {
        const int buf = s & 1;
        if (s + 1 < STAGES) {
            issue(buf ^ 1, (s + 1) * BK);
            asm volatile("s_wait_asynccnt 0x4" ::: "memory");  // stage s landed
        } else {
            asm volatile("s_wait_asynccnt 0x0" ::: "memory");
        }
        __syncthreads();   // stage s visible to all waves

        BF16x16 af[2];
#pragma unroll
        for (int mf = 0; mf < 2; ++mf) {
            const __hip_bfloat16* p = &As[buf][waveM * 32 + mf * 16 + laneM][laneK];
            af[mf].q[0] = *(const uint4*)p;
            af[mf].q[1] = *(const uint4*)(p + 16);
        }
#pragma unroll
        for (int nf = 0; nf < 4; ++nf) {
            BF16x16 bfr;
            const __hip_bfloat16* p = &Bs[buf][waveN * 64 + nf * 16 + laneM][laneK];
            bfr.q[0] = *(const uint4*)p;
            bfr.q[1] = *(const uint4*)(p + 16);
#pragma unroll
            for (int mf = 0; mf < 2; ++mf)
                acc[mf][nf] = __builtin_amdgcn_wmma_f32_16x16x32_bf16(
                    false, af[mf].v, false, bfr.v, (short)0, acc[mf][nf], false, false);
        }
        __syncthreads();   // done reading buf before it is overwritten at s+1's issue
    }

    const int rowHi = (lane >> 4) << 3;   // C/D: VGPR r -> M=r (lanes 0-15) / r+8 (16-31)
#pragma unroll
    for (int mf = 0; mf < 2; ++mf)
#pragma unroll
        for (int nf = 0; nf < 4; ++nf)
#pragma unroll
            for (int r = 0; r < 8; ++r)
                C[(size_t)(mBase + waveM * 32 + mf * 16 + r + rowHi) * Hout +
                  (nBase + waveN * 64 + nf * 16 + laneM)] = acc[mf][nf][r];
}

// ---------------- aggregation ----------------
__global__ void k_agg_self(const float* __restrict__ t, const float* __restrict__ dinv,
                           float* __restrict__ a, int H, size_t total) {
    size_t i = (size_t)blockIdx.x * blockDim.x + threadIdx.x;
    if (i < total) {
        int n = (int)(i / H);
        float d = dinv[n];
        a[i] = t[i] * d * d;
    }
}

__global__ void k_agg_edges(const int* __restrict__ src, const int* __restrict__ dst,
                            const float* __restrict__ norm, const float* __restrict__ t,
                            float* __restrict__ a) {
    int idx = blockIdx.x * blockDim.x + threadIdx.x;
    int e = idx >> 7;                  // HDIM/4 = 128 chunks per edge
    if (e >= E_EDGES) return;
    int f = (idx & 127) << 2;
    float nm = norm[e];
    const float4 v = *(const float4*)(t + (size_t)src[e] * HDIM + f);
    float* o = a + (size_t)dst[e] * HDIM + f;
    unsafeAtomicAdd(o + 0, nm * v.x);
    unsafeAtomicAdd(o + 1, nm * v.y);
    unsafeAtomicAdd(o + 2, nm * v.z);
    unsafeAtomicAdd(o + 3, nm * v.w);
}

__global__ void k_agg_edges_c2(const int* __restrict__ src, const int* __restrict__ dst,
                               const float* __restrict__ norm, const float* __restrict__ t4,
                               float* __restrict__ a4) {
    int e = blockIdx.x * blockDim.x + threadIdx.x;
    if (e >= E_EDGES) return;
    float nm = norm[e];
    const float* v = t4 + (size_t)src[e] * CDIM;
    float* o = a4 + (size_t)dst[e] * CDIM;
    unsafeAtomicAdd(o + 0, nm * v[0]);
    unsafeAtomicAdd(o + 1, nm * v[1]);
}

__global__ void k_bias_relu_cast(const float* __restrict__ a, const float* __restrict__ bias,
                                 __hip_bfloat16* __restrict__ h, size_t total) {
    size_t i = (size_t)blockIdx.x * blockDim.x + threadIdx.x;
    if (i < total) {
        int f = (int)(i % HDIM);
        float v = a[i] + bias[f];
        h[i] = __float2bfloat16(v > 0.0f ? v : 0.0f);
    }
}

// ---------------- tiny layer 4 + log-softmax ----------------
__global__ void k_gemm4(const __hip_bfloat16* __restrict__ h, const float* __restrict__ W4,
                        float* __restrict__ t4) {
    int n = blockIdx.x * blockDim.x + threadIdx.x;
    if (n >= N_NODES) return;
    const __hip_bfloat16* row = h + (size_t)n * HDIM;
    float a0 = 0.0f, a1 = 0.0f;
    for (int k = 0; k < HDIM; ++k) {
        float v = __bfloat162float(row[k]);
        a0 += v * W4[k * 2 + 0];
        a1 += v * W4[k * 2 + 1];
    }
    t4[n * 2 + 0] = a0;
    t4[n * 2 + 1] = a1;
}

__global__ void k_logsoftmax(const float* __restrict__ a4, const float* __restrict__ b4,
                             float* __restrict__ out) {
    int n = blockIdx.x * blockDim.x + threadIdx.x;
    if (n >= N_NODES) return;
    float z0 = a4[n * 2 + 0] + b4[0];
    float z1 = a4[n * 2 + 1] + b4[1];
    float m = fmaxf(z0, z1);
    float l = m + logf(expf(z0 - m) + expf(z1 - m));
    out[n * 2 + 0] = z0 - l;
    out[n * 2 + 1] = z1 - l;
}

// ---------------- host-side orchestration ----------------
extern "C" void kernel_launch(void* const* d_in, const int* in_sizes, int n_in,
                              void* d_out, int out_size, void* d_ws, size_t ws_size,
                              hipStream_t stream) {
    const float* x  = (const float*)d_in[0];
    const int*   ei = (const int*)d_in[1];    // [2, E]
    const float* W1 = (const float*)d_in[3];
    const float* b1 = (const float*)d_in[4];
    const float* W2 = (const float*)d_in[5];
    const float* b2 = (const float*)d_in[6];
    const float* W3 = (const float*)d_in[7];
    const float* b3 = (const float*)d_in[8];
    const float* W4 = (const float*)d_in[9];
    const float* b4 = (const float*)d_in[10];
    float* out = (float*)d_out;

    const int* src = ei;
    const int* dst = ei + E_EDGES;

    // workspace carve-up (256B aligned); GEMM operands padded to MPAD rows
    char* ws = (char*)d_ws;
    size_t off = 0;
    auto carve = [&](size_t bytes) {
        void* p = ws + off;
        off = (off + bytes + 255) & ~(size_t)255;
        return p;
    };
    float*          deg  = (float*)carve((size_t)N_NODES * 4);
    float*          dinv = (float*)carve((size_t)N_NODES * 4);
    float*          norm = (float*)carve((size_t)E_EDGES * 4);
    __hip_bfloat16* xb   = (__hip_bfloat16*)carve((size_t)MPAD * F_INDIM * 2);
    __hip_bfloat16* W1t  = (__hip_bfloat16*)carve((size_t)HDIM * F_INDIM * 2);
    __hip_bfloat16* W2t  = (__hip_bfloat16*)carve((size_t)HDIM * HDIM * 2);
    __hip_bfloat16* W3t  = (__hip_bfloat16*)carve((size_t)HDIM * HDIM * 2);
    float*          t    = (float*)carve((size_t)MPAD * HDIM * 4);
    float*          a    = (float*)carve((size_t)N_NODES * HDIM * 4);
    __hip_bfloat16* h    = (__hip_bfloat16*)carve((size_t)MPAD * HDIM * 2);
    float*          t4   = (float*)carve((size_t)N_NODES * CDIM * 4);
    float*          a4   = (float*)carve((size_t)N_NODES * CDIM * 4);
    (void)ws_size; (void)in_sizes; (void)n_in; (void)out_size;

    const int T = 256;
    auto blk = [](size_t n, int t) { return (int)((n + t - 1) / t); };

    // graph norm
    k_deg_init<<<blk(N_NODES, T), T, 0, stream>>>(deg);
    k_deg_edges<<<blk(E_EDGES, T), T, 0, stream>>>(dst, deg);
    k_dinv<<<blk(N_NODES, T), T, 0, stream>>>(deg, dinv);
    k_norm<<<blk(E_EDGES, T), T, 0, stream>>>(src, dst, dinv, norm);

    // precision prep (pad rows of xb/h are never consumed downstream)
    k_cast_bf16<<<blk((size_t)N_NODES * F_INDIM, T), T, 0, stream>>>(x, xb, (size_t)N_NODES * F_INDIM);
    k_transpose_bf16<<<blk((size_t)F_INDIM * HDIM, T), T, 0, stream>>>(W1, W1t, F_INDIM, HDIM);
    k_transpose_bf16<<<blk((size_t)HDIM * HDIM, T), T, 0, stream>>>(W2, W2t, HDIM, HDIM);
    k_transpose_bf16<<<blk((size_t)HDIM * HDIM, T), T, 0, stream>>>(W3, W3t, HDIM, HDIM);

    const size_t NH = (size_t)N_NODES * HDIM;
    dim3 gGrid(HDIM / BN, MPAD / BM);   // 4 x 79

    // layer 1
    k_gemm_lds<F_INDIM><<<gGrid, 256, 0, stream>>>(xb, W1t, t, HDIM);
    k_agg_self<<<blk(NH, T), T, 0, stream>>>(t, dinv, a, HDIM, NH);
    k_agg_edges<<<blk((size_t)E_EDGES * 128, T), T, 0, stream>>>(src, dst, norm, t, a);
    k_bias_relu_cast<<<blk(NH, T), T, 0, stream>>>(a, b1, h, NH);

    // layer 2
    k_gemm_lds<HDIM><<<gGrid, 256, 0, stream>>>(h, W2t, t, HDIM);
    k_agg_self<<<blk(NH, T), T, 0, stream>>>(t, dinv, a, HDIM, NH);
    k_agg_edges<<<blk((size_t)E_EDGES * 128, T), T, 0, stream>>>(src, dst, norm, t, a);
    k_bias_relu_cast<<<blk(NH, T), T, 0, stream>>>(a, b2, h, NH);

    // layer 3
    k_gemm_lds<HDIM><<<gGrid, 256, 0, stream>>>(h, W3t, t, HDIM);
    k_agg_self<<<blk(NH, T), T, 0, stream>>>(t, dinv, a, HDIM, NH);
    k_agg_edges<<<blk((size_t)E_EDGES * 128, T), T, 0, stream>>>(src, dst, norm, t, a);
    k_bias_relu_cast<<<blk(NH, T), T, 0, stream>>>(a, b3, h, NH);

    // layer 4 + log-softmax
    k_gemm4<<<blk(N_NODES, T), T, 0, stream>>>(h, W4, t4);
    k_agg_self<<<blk((size_t)N_NODES * CDIM, T), T, 0, stream>>>(t4, dinv, a4, CDIM, (size_t)N_NODES * CDIM);
    k_agg_edges_c2<<<blk(E_EDGES, T), T, 0, stream>>>(src, dst, norm, t4, a4);
    k_logsoftmax<<<blk(N_NODES, T), T, 0, stream>>>(a4, b4, out);
}
